// AttnDecoder_57492432224734
// MI455X (gfx1250) — matrix operand order repaired and checked by reading, not compiled, not run on hardware
//
#include <hip/hip_runtime.h>
#include <hip/hip_bf16.h>
#include <math.h>

// Problem constants from the reference
#define BB 32
#define SS 512
#define HH 1024
#define VV 8192

typedef __attribute__((ext_vector_type(16))) __bf16 v16bf;
typedef __attribute__((ext_vector_type(8)))  __bf16 v8bf;
typedef __attribute__((ext_vector_type(8)))  float  v8f;

__device__ __forceinline__ unsigned short f2bf(float f) {
  unsigned int u = __float_as_uint(f);
  unsigned int r = (u + 0x7FFFu + ((u >> 16) & 1u)) >> 16;   // round-to-nearest-even
  return (unsigned short)r;
}

__device__ __forceinline__ float sigf(float x) { return 1.0f / (1.0f + expf(-x)); }

// ---------------------------------------------------------------------------
// Preprocessing: f32 -> bf16 weight packing
// ---------------------------------------------------------------------------
__global__ __launch_bounds__(256) void k_cvt(const float* __restrict__ src,
                                             unsigned short* __restrict__ dst, int n) {
  int i = blockIdx.x * 256 + threadIdx.x;
  if (i < n) dst[i] = f2bf(src[i]);
}

// Wg[r][c] : c<2048 -> W_ih[r][c], else W_hh[r][c-2048]   (r < 4096, c < 3072)
__global__ __launch_bounds__(256) void k_pack_gw(const float* __restrict__ wih,
                                                 const float* __restrict__ whh,
                                                 unsigned short* __restrict__ wg) {
  int i = blockIdx.x * 256 + threadIdx.x;
  int r = i / 3072, c = i - r * 3072;
  float v = (c < 2048) ? wih[(size_t)r * 2048 + c] : whh[(size_t)r * 1024 + (c - 2048)];
  wg[i] = f2bf(v);
}

__global__ __launch_bounds__(256) void k_bias2(const float* __restrict__ a,
                                               const float* __restrict__ b,
                                               float* __restrict__ o) {
  int i = blockIdx.x * 256 + threadIdx.x;
  o[i] = a[i] + b[i];
}

// ---------------------------------------------------------------------------
// Per-step: build X = [enc[:,s,:] | ctxvec[:,s,:] | h] in bf16   ([32, 3072])
// ---------------------------------------------------------------------------
__global__ __launch_bounds__(256) void k_pack_x(const float* __restrict__ enc,
                                                const float* __restrict__ cvec,
                                                const float* __restrict__ h,
                                                unsigned short* __restrict__ X, int s) {
  int i = blockIdx.x * 256 + threadIdx.x;          // < 32*3072
  int b = i / 3072, c = i - b * 3072;
  float v;
  if (c < 1024)       v = enc[((size_t)b * SS + s) * HH + c];
  else if (c < 2048)  v = cvec[((size_t)b * SS + s) * HH + (c - 1024)];
  else                v = h[b * HH + (c - 2048)];
  X[i] = f2bf(v);
}

// ---------------------------------------------------------------------------
// WMMA GEMM:  out[32,N] = A[32,K](bf16) @ Bw[N,K](bf16)^T  (+bias)(+tanh)
// Each wave: 32x32 output block (2 M-tiles x 2 N-tiles), block = 8 waves = 256 N
// ---------------------------------------------------------------------------
__global__ __launch_bounds__(256) void k_gemm(const __bf16* __restrict__ A,
                                              const __bf16* __restrict__ Bw,
                                              const float* __restrict__ bias,
                                              float* __restrict__ outF,
                                              unsigned short* __restrict__ outB,
                                              int K, int N, int act) {
  const int lane  = threadIdx.x & 31;
  const int wave  = threadIdx.x >> 5;
  const int n0    = blockIdx.x * 256 + wave * 32;
  const int l15   = lane & 15;
  const int khalf = lane >> 4;

  // A: 16-bit 16x32 layout — lane m=l15; half-wave picks K 0..7/16..23 vs 8..15/24..31
  const __bf16* pa0 = A + (size_t)l15 * K + khalf * 8;
  const __bf16* pa1 = A + (size_t)(l15 + 16) * K + khalf * 8;
  // B: 16-bit 32x16 layout — lane n=l15; half-wave picks K 0..7/16..23 vs 8..15/24..31
  const __bf16* pb0 = Bw + (size_t)(n0 + l15) * K + khalf * 8;
  const __bf16* pb1 = Bw + (size_t)(n0 + 16 + l15) * K + khalf * 8;

  v8f acc[2][2] = {};

  for (int k = 0; k < K; k += 32) {
    union { v16bf v; v8bf h[2]; } a0, a1, b0, b1;
    a0.h[0] = *(const v8bf*)(pa0 + k);
    a0.h[1] = *(const v8bf*)(pa0 + k + 16);
    a1.h[0] = *(const v8bf*)(pa1 + k);
    a1.h[1] = *(const v8bf*)(pa1 + k + 16);
    b0.h[0] = *(const v8bf*)(pb0 + k);
    b0.h[1] = *(const v8bf*)(pb0 + k + 16);
    b1.h[0] = *(const v8bf*)(pb1 + k);
    b1.h[1] = *(const v8bf*)(pb1 + k + 16);
    acc[0][0] = __builtin_amdgcn_wmma_f32_16x16x32_bf16(false, a0.v, false, b0.v,
                                                        (short)0, acc[0][0], false, false);
    acc[0][1] = __builtin_amdgcn_wmma_f32_16x16x32_bf16(false, a0.v, false, b1.v,
                                                        (short)0, acc[0][1], false, false);
    acc[1][0] = __builtin_amdgcn_wmma_f32_16x16x32_bf16(false, a1.v, false, b0.v,
                                                        (short)0, acc[1][0], false, false);
    acc[1][1] = __builtin_amdgcn_wmma_f32_16x16x32_bf16(false, a1.v, false, b1.v,
                                                        (short)0, acc[1][1], false, false);
  }

  // D layout: lanes 0-15 -> M = vg, lanes 16-31 -> M = 8+vg ; N = l15 (+16 per n-tile)
  const int mbase = khalf * 8;
#pragma unroll
  for (int mt = 0; mt < 2; ++mt) {
#pragma unroll
    for (int nt = 0; nt < 2; ++nt) {
      int col = n0 + nt * 16 + l15;
      float bv = bias ? bias[col] : 0.0f;
#pragma unroll
      for (int vg = 0; vg < 8; ++vg) {
        int row = mt * 16 + mbase + vg;
        float x = acc[mt][nt][vg] + bv;
        if (act) x = tanhf(x);
        size_t idx = (size_t)row * N + col;
        if (outF) outF[idx] = x;
        if (outB) outB[idx] = f2bf(x);
      }
    }
  }
}

// ---------------------------------------------------------------------------
// LSTM elementwise update: gates [32,4096] -> c,h (f32) and h (bf16)
// ---------------------------------------------------------------------------
__global__ __launch_bounds__(256) void k_lstm(const float* __restrict__ g,
                                              float* __restrict__ c,
                                              float* __restrict__ h,
                                              unsigned short* __restrict__ hbf) {
  int i = blockIdx.x * 256 + threadIdx.x;          // < 32*1024
  int b = i >> 10, j = i & 1023;
  const float* gr = g + (size_t)b * 4096;
  float ig = gr[j], fg = gr[j + 1024], gg = gr[j + 2048], og = gr[j + 3072];
  float cn = sigf(fg) * c[i] + sigf(ig) * tanhf(gg);
  float hn = sigf(og) * tanhf(cn);
  c[i] = cn;
  h[i] = hn;
  hbf[i] = f2bf(hn);
}

// ---------------------------------------------------------------------------
// Attention: scores -> masked softmax -> context; emit cat=[ctx|h] bf16 [32,2048]
// One block per batch row.
// ---------------------------------------------------------------------------
__global__ __launch_bounds__(256) void k_attn(const float* __restrict__ enc,
                                              const float* __restrict__ q,
                                              const int* __restrict__ xs_len,
                                              const float* __restrict__ h,
                                              unsigned short* __restrict__ cat) {
  __shared__ float sc[SS];
  __shared__ float red[256];
  const int b = blockIdx.x, t = threadIdx.x;
  const float* qb = q + b * HH;
  const float* eb = enc + (size_t)b * SS * HH;
  const int len = xs_len[b];

  for (int s = t; s < SS; s += 256) {
    const float4* e4 = (const float4*)(eb + (size_t)s * HH);
    const float4* q4 = (const float4*)qb;
    float d = 0.0f;
    for (int i = 0; i < HH / 4; ++i) {
      float4 a = e4[i], cc = q4[i];
      d += a.x * cc.x + a.y * cc.y + a.z * cc.z + a.w * cc.w;
    }
    float m = (s < len) ? d : 0.0f;                // mask*score
    sc[s] = (m == 0.0f) ? -1e10f : m;              // where(m==0, -1e10, m)
  }
  __syncthreads();

  float mx = -INFINITY;
  for (int s = t; s < SS; s += 256) mx = fmaxf(mx, sc[s]);
  red[t] = mx; __syncthreads();
  for (int o = 128; o > 0; o >>= 1) { if (t < o) red[t] = fmaxf(red[t], red[t + o]); __syncthreads(); }
  mx = red[0]; __syncthreads();

  float ls = 0.0f;
  for (int s = t; s < SS; s += 256) { float e = expf(sc[s] - mx); sc[s] = e; ls += e; }
  red[t] = ls; __syncthreads();
  for (int o = 128; o > 0; o >>= 1) { if (t < o) red[t] += red[t + o]; __syncthreads(); }
  float inv = 1.0f / red[0]; __syncthreads();

  for (int hh = t; hh < HH; hh += 256) {
    float a = 0.0f;
    for (int s = 0; s < SS; ++s) a += sc[s] * eb[(size_t)s * HH + hh];
    a *= inv;
    cat[(size_t)b * 2048 + hh]        = f2bf(a);
    cat[(size_t)b * 2048 + 1024 + hh] = f2bf(h[b * HH + hh]);
  }
}

// ---------------------------------------------------------------------------
// log_softmax over V and write out[b, s, :]. One block per batch row.
// ---------------------------------------------------------------------------
__global__ __launch_bounds__(256) void k_lsm(const float* __restrict__ sc,
                                             float* __restrict__ out, int s) {
  __shared__ float red[256];
  const int b = blockIdx.x, t = threadIdx.x;
  const float* r = sc + (size_t)b * VV;
  float mx = -INFINITY;
  for (int v = t; v < VV; v += 256) mx = fmaxf(mx, r[v]);
  red[t] = mx; __syncthreads();
  for (int o = 128; o > 0; o >>= 1) { if (t < o) red[t] = fmaxf(red[t], red[t + o]); __syncthreads(); }
  mx = red[0]; __syncthreads();
  float ls = 0.0f;
  for (int v = t; v < VV; v += 256) ls += expf(r[v] - mx);
  red[t] = ls; __syncthreads();
  for (int o = 128; o > 0; o >>= 1) { if (t < o) red[t] += red[t + o]; __syncthreads(); }
  float lse = mx + logf(red[0]);
  float* ob = out + ((size_t)b * SS + s) * VV;
  for (int v = t; v < VV; v += 256) ob[v] = r[v] - lse;
}

// ---------------------------------------------------------------------------
extern "C" void kernel_launch(void* const* d_in, const int* in_sizes, int n_in,
                              void* d_out, int out_size, void* d_ws, size_t ws_size,
                              hipStream_t stream) {
  const float* enc   = (const float*)d_in[0];
  const float* h0    = (const float*)d_in[1];
  const float* c0    = (const float*)d_in[2];
  const float* cvec  = (const float*)d_in[3];
  const float* W_ih  = (const float*)d_in[4];
  const float* b_ih  = (const float*)d_in[5];
  const float* W_hh  = (const float*)d_in[6];
  const float* b_hh  = (const float*)d_in[7];
  const float* a1w   = (const float*)d_in[8];
  const float* a2w   = (const float*)d_in[9];
  const float* l1w   = (const float*)d_in[10];
  const float* l1b   = (const float*)d_in[11];
  const float* l2w   = (const float*)d_in[12];
  const float* l2b   = (const float*)d_in[13];
  const int*   xlen  = (const int*)d_in[14];
  float* out = (float*)d_out;

  size_t off = 0;
  auto alloc = [&](size_t bytes) -> void* {
    void* p = (char*)d_ws + off;
    off += (bytes + 255) & ~(size_t)255;
    return p;
  };
  unsigned short* Wg    = (unsigned short*)alloc((size_t)4096 * 3072 * 2);
  unsigned short* A1    = (unsigned short*)alloc((size_t)1024 * 1024 * 2);
  unsigned short* A2    = (unsigned short*)alloc((size_t)1024 * 2048 * 2);
  unsigned short* L1    = (unsigned short*)alloc((size_t)1024 * 1024 * 2);
  unsigned short* L2    = (unsigned short*)alloc((size_t)8192 * 1024 * 2);
  float*          bg    = (float*)alloc(4096 * 4);
  float*          hbuf  = (float*)alloc(BB * HH * 4);
  float*          cbuf  = (float*)alloc(BB * HH * 4);
  unsigned short* hbf   = (unsigned short*)alloc(BB * HH * 2);
  unsigned short* Xbf   = (unsigned short*)alloc(BB * 3072 * 2);
  float*          gates = (float*)alloc(BB * 4096 * 4);
  float*          qf    = (float*)alloc(BB * HH * 4);
  unsigned short* catbf = (unsigned short*)alloc(BB * 2048 * 2);
  unsigned short* c2bf  = (unsigned short*)alloc(BB * HH * 2);
  unsigned short* t1bf  = (unsigned short*)alloc(BB * HH * 2);
  float*          scf   = (float*)alloc((size_t)BB * VV * 4);

  // ---- preprocessing: bf16 weights (L2-resident working set ~50MB) ----
  k_pack_gw<<<(4096 * 3072) / 256, 256, 0, stream>>>(W_ih, W_hh, Wg);
  k_cvt<<<(1024 * 1024) / 256, 256, 0, stream>>>(a1w, A1, 1024 * 1024);
  k_cvt<<<(1024 * 2048) / 256, 256, 0, stream>>>(a2w, A2, 1024 * 2048);
  k_cvt<<<(1024 * 1024) / 256, 256, 0, stream>>>(l1w, L1, 1024 * 1024);
  k_cvt<<<(8192 * 1024) / 256, 256, 0, stream>>>(l2w, L2, 8192 * 1024);
  k_bias2<<<4096 / 256, 256, 0, stream>>>(b_ih, b_hh, bg);
  hipMemcpyAsync(hbuf, h0, (size_t)BB * HH * 4, hipMemcpyDeviceToDevice, stream);
  hipMemcpyAsync(cbuf, c0, (size_t)BB * HH * 4, hipMemcpyDeviceToDevice, stream);

  // ---- sequential scan over S (serialized by stream order) ----
  for (int s = 0; s < SS; ++s) {
    k_pack_x<<<(BB * 3072) / 256, 256, 0, stream>>>(enc, cvec, hbuf, Xbf, s);
    // gates = [e|cv|h] @ [W_ih|W_hh]^T + (b_ih+b_hh)
    k_gemm<<<4096 / 256, 256, 0, stream>>>((const __bf16*)Xbf, (const __bf16*)Wg,
                                           bg, gates, nullptr, 3072, 4096, 0);
    k_lstm<<<(BB * HH) / 256, 256, 0, stream>>>(gates, cbuf, hbuf, hbf);
    // q = h_new @ attn_l1_w^T
    k_gemm<<<1024 / 256, 256, 0, stream>>>((const __bf16*)hbf, (const __bf16*)A1,
                                           nullptr, qf, nullptr, 1024, 1024, 0);
    k_attn<<<BB, 256, 0, stream>>>(enc, qf, xlen, hbuf, catbf);
    // ctx2 = tanh([ctx|h] @ attn_l2_w^T)
    k_gemm<<<1024 / 256, 256, 0, stream>>>((const __bf16*)catbf, (const __bf16*)A2,
                                           nullptr, nullptr, c2bf, 2048, 1024, 1);
    // t1 = tanh(ctx2 @ lin1_w^T + lin1_b)
    k_gemm<<<1024 / 256, 256, 0, stream>>>((const __bf16*)c2bf, (const __bf16*)L1,
                                           l1b, nullptr, t1bf, 1024, 1024, 1);
    // sc = t1 @ lin2_w^T + lin2_b
    k_gemm<<<8192 / 256, 256, 0, stream>>>((const __bf16*)t1bf, (const __bf16*)L2,
                                           l2b, scf, nullptr, 1024, 8192, 0);
    k_lsm<<<BB, 256, 0, stream>>>(scf, out, s);
  }
}
